// MultiHeadedSelfAttention_86870008529585
// MI455X (gfx1250) — matrix operand order repaired and checked by reading, not compiled, not run on hardware
//
#include <hip/hip_runtime.h>
#include <hip/hip_bf16.h>

// ---------------- CDNA5 WMMA types ----------------
typedef __attribute__((ext_vector_type(16))) _Float16 v16h;
typedef __attribute__((ext_vector_type(8)))  float    v8f;
typedef __attribute__((ext_vector_type(4)))  unsigned int u4;
typedef __attribute__((ext_vector_type(8)))  int      i8v;
typedef __attribute__((ext_vector_type(4)))  int      i4v;
typedef __attribute__((ext_vector_type(4)))  float    f4;

union F16x16 { v16h v; u4 q[2]; _Float16 h[16]; };
union VF8    { v8f  v; float f[8]; };

struct PkHalf { u4 q0, q1; };   // 16 packed f16

static __device__ __forceinline__ v8f wmma_f16(const v16h& a, const v16h& b, v8f c) {
    // D = A(16x32 f16) * B(32x16 f16) + C(16x16 f32)
    return __builtin_amdgcn_wmma_f32_16x16x32_f16(false, a, false, b, (short)0, c, false, false);
}

static __device__ __forceinline__ PkHalf cvt_f32x16(const float* __restrict__ p) {
    union { u4 q[2]; _Float16 h[16]; } pk;
#pragma unroll
    for (int i = 0; i < 4; ++i) {
        f4 f = *(const f4*)(p + 4 * i);
        pk.h[4*i+0] = (_Float16)f.x; pk.h[4*i+1] = (_Float16)f.y;
        pk.h[4*i+2] = (_Float16)f.z; pk.h[4*i+3] = (_Float16)f.w;
    }
    return { pk.q[0], pk.q[1] };
}

// LDS byte offset of a __shared__ object (addrspace(3) ptrtoint)
static __device__ __forceinline__ unsigned lds_offset(const void* p) {
    typedef __attribute__((address_space(3))) const char lds_char;
    return (unsigned)(unsigned long long)(lds_char*)(p);
}

#define D_MODEL 1024
#define SEQ     2048
#define BATCH   4
#define HEADS   16
#define DK      64

// =====================================================================
// TDM: DMA one 128x32-half f16 tile (row stride 1024 elems in memory)
// into LDS with row stride 40 halves (pad 4 DWORDs every 16 DWORDs).
// Issued by one wave per block; completion via s_wait_tensorcnt.
// =====================================================================
static __device__ __forceinline__ void tdm_load_tile_f16(
    const _Float16* gsrc, unsigned lds_byte_addr)
{
    const unsigned long long ga = (unsigned long long)(size_t)gsrc;
    u4 g0;
    g0[0] = 1u;                                         // count=1 (user D#)
    g0[1] = lds_byte_addr;                              // lds_addr
    g0[2] = (unsigned)(ga & 0xFFFFFFFFu);               // global_addr[31:0]
    g0[3] = (unsigned)((ga >> 32) & 0x01FFFFFFu)        // global_addr[56:32]
            | (2u << 30);                               // type=2 ("image")
    i8v g1;
    g1[0] = (int)((1u << 16)      // data_size = 1 -> 2 bytes
                | (1u << 20)      // pad_enable
                | (3u << 22)      // pad_interval: code 3 = 16 DWORDs
                | (3u << 25));    // pad_amount:   code 3 = 4 DWORDs
    g1[1] = (int)(1024u << 16);                 // tensor_dim0 = 1024 (lo16)
    g1[2] = (int)(8192u << 16);                 // dim0 hi16=0 | tensor_dim1 lo16
    g1[3] = (int)(32u << 16);                   // dim1 hi16=0 | tile_dim0 = 32
    g1[4] = (int)128u;                          // tile_dim1 = 128, tile_dim2 = 0
    g1[5] = (int)1024u;                         // tensor_dim0_stride lo32
    g1[6] = 0;                                  // stride0 hi | stride1 lo
    g1[7] = 0;
    const i4v z4 = { 0, 0, 0, 0 };
    const i8v z8 = { 0, 0, 0, 0, 0, 0, 0, 0 };
    // 6-arg form (this toolchain): (g0, g1, g2, g3, g4, cpol)
    __builtin_amdgcn_tensor_load_to_lds(g0, g1, z4, z4, z8, 0);
}

// =====================================================================
// Tiled WMMA GEMM:  C[m,n] = sum_k A[m,k] * W[n,k] + bias[n]
// Double-buffered LDS pipeline; AF16/OUTMODE compile-time.
// AF16==1: A tile staged by the Tensor Data Mover (async DMA to LDS).
// OUTMODE 0: f16 -> [B,H,S,DK]   (Q / K)
// OUTMODE 1: f16 -> [B,H,DK,S]   (V pre-transposed for PV)
// OUTMODE 2: f32 -> [8192 x 1024] (final projection)
// Block: 256 threads = 8 waves; block tile 128x128; wave tile 32x64.
// =====================================================================
template<int AF16, int OUTMODE>
__global__ __launch_bounds__(256) void gemm_wmma_kernel(
    const void* __restrict__ Aptr,
    const float* __restrict__ W, const float* __restrict__ bias,
    void* __restrict__ Out)
{
    __shared__ __align__(16) _Float16 As[2][128][40];   // 2 x 10 KB
    __shared__ __align__(16) _Float16 Bs[2][128][40];   // 2 x 10 KB

    const int tid   = threadIdx.x;
    const int wave  = tid >> 5;
    const int lane  = tid & 31;
    const int lm    = lane & 15;
    const int sel   = lane >> 4;
    const int wm    = wave & 3;        // wave row 0..3  (32 rows each)
    const int wn    = wave >> 2;       // wave col 0..1  (64 cols each)
    const int Mbase = blockIdx.x * 128;
    const int Nbase = blockIdx.y * 128;

    const int lrow = tid >> 1;         // 0..127 : tile row this thread stages
    const int lseg = (tid & 1) * 16;   // 0 or 16: 16 K-elements per thread

    const size_t arowbase = (size_t)(Mbase + lrow) * D_MODEL + lseg;
    const size_t browbase = (size_t)(Nbase + lrow) * D_MODEL + lseg;

    VF8 acc[2][4];
#pragma unroll
    for (int i = 0; i < 2; ++i)
#pragma unroll
        for (int j = 0; j < 4; ++j)
#pragma unroll
            for (int r = 0; r < 8; ++r) acc[i][j].f[r] = 0.0f;

    auto loadB = [&](int k0) -> PkHalf { return cvt_f32x16(W + browbase + k0); };
    auto stageA_tdm = [&](int buf, int k0) {           // AF16==1 path
        if (wave == 0)
            tdm_load_tile_f16((const _Float16*)Aptr + (size_t)Mbase * D_MODEL + k0,
                              lds_offset(&As[buf][0][0]));
    };

    // ---- prologue: stage tile 0 ----
    PkHalf pa;
    if constexpr (AF16 == 0) {
        pa = cvt_f32x16((const float*)Aptr + arowbase);
        _Float16* da = &As[0][lrow][lseg];
        *(u4*)da = pa.q0; *((u4*)da + 1) = pa.q1;
    } else {
        stageA_tdm(0, 0);
    }
    PkHalf pb = loadB(0);
    {
        _Float16* db = &Bs[0][lrow][lseg];
        *(u4*)db = pb.q0; *((u4*)db + 1) = pb.q1;
    }
    if constexpr (AF16 == 1) {
        if (wave == 0) __builtin_amdgcn_s_wait_tensorcnt(0);
    }
    __syncthreads();

    constexpr int NK = D_MODEL / 32;   // 32 k-steps
#pragma unroll 2
    for (int kk = 0; kk < NK; ++kk) {
        const int cur = kk & 1;
        // stage next tile (global->regs, or TDM DMA->LDS) while WMMAs run
        if (kk + 1 < NK) {
            if constexpr (AF16 == 0) {
                pa = cvt_f32x16((const float*)Aptr + arowbase + (kk + 1) * 32);
            } else {
                stageA_tdm(cur ^ 1, (kk + 1) * 32);
            }
            pb = loadB((kk + 1) * 32);
        }
        // ---- fragment loads per ISA 7.12.2 layouts ----
        F16x16 af[2];
#pragma unroll
        for (int i = 0; i < 2; ++i) {
            const _Float16* p = &As[cur][wm * 32 + i * 16 + lm][sel * 8];
            af[i].q[0] = *(const u4*)(p);        // K = sel*8 .. +7
            af[i].q[1] = *(const u4*)(p + 16);   // K = 16+sel*8 .. +7
        }
        F16x16 bf[4];
#pragma unroll
        for (int j = 0; j < 4; ++j) {
            const _Float16* p = &Bs[cur][wn * 64 + j * 16 + lm][sel * 8];
            bf[j].q[0] = *(const u4*)(p);        // K = sel*8 .. +7
            bf[j].q[1] = *(const u4*)(p + 16);   // K = 16+sel*8 .. +7
        }
#pragma unroll
        for (int i = 0; i < 2; ++i)
#pragma unroll
            for (int j = 0; j < 4; ++j)
                acc[i][j].v = wmma_f16(af[i].v, bf[j].v, acc[i][j].v);

        if (kk + 1 < NK) {
            const int nxt = cur ^ 1;
            if constexpr (AF16 == 0) {
                _Float16* da = &As[nxt][lrow][lseg];
                *(u4*)da = pa.q0; *((u4*)da + 1) = pa.q1;
            }
            _Float16* db = &Bs[nxt][lrow][lseg];
            *(u4*)db = pb.q0; *((u4*)db + 1) = pb.q1;
            if constexpr (AF16 == 1) {
                if (wave == 0) __builtin_amdgcn_s_wait_tensorcnt(0);
            }
            __syncthreads();   // one barrier per k-step (ping-pong)
        }
    }

    // ---- epilogue: C frag row m = r + 8*sel, col n = lane&15 ----
#pragma unroll
    for (int i = 0; i < 2; ++i) {
#pragma unroll
        for (int j = 0; j < 4; ++j) {
            const int mrow0 = Mbase + wm * 32 + i * 16;
            const int ncol  = Nbase + wn * 64 + j * 16 + lm;
            const float bv  = bias[ncol];
            if constexpr (OUTMODE == 0) {            // f16 [B,H,S,DK]
                const int h = ncol >> 6, dd = ncol & 63;
#pragma unroll
                for (int r = 0; r < 8; ++r) {
                    const int m = mrow0 + r + 8 * sel;
                    const int b = m >> 11, s = m & (SEQ - 1);
                    _Float16* dst = (_Float16*)Out +
                        ((((size_t)b * HEADS + h) * SEQ + s) * DK + dd);
                    *dst = (_Float16)(acc[i][j].f[r] + bv);
                }
            } else if constexpr (OUTMODE == 1) {     // f16 [B,H,DK,S] (V^T)
                const int h = ncol >> 6, dd = ncol & 63;
                const int m0 = mrow0 + 8 * sel;      // 8 contiguous tokens
                const int b = m0 >> 11, s0 = m0 & (SEQ - 1);
                union { u4 q; _Float16 h8[8]; } pk;
#pragma unroll
                for (int r = 0; r < 8; ++r) pk.h8[r] = (_Float16)(acc[i][j].f[r] + bv);
                _Float16* dst = (_Float16*)Out +
                    ((((size_t)b * HEADS + h) * DK + dd) * SEQ + s0);
                *(u4*)dst = pk.q;                    // one 16B store per frag
            } else {                                 // f32 [8192 x 1024]
#pragma unroll
                for (int r = 0; r < 8; ++r) {
                    const int m = mrow0 + r + 8 * sel;
                    ((float*)Out)[(size_t)m * D_MODEL + ncol] = acc[i][j].f[r] + bv;
                }
            }
        }
    }
}

// =====================================================================
// Flash-style attention per (b,h): O = softmax(mask(Q K^T / 8)) V
// Q,K: f16 [B,H,S,DK]; Vt: f16 [B,H,DK,S]; Ows: f16 [B,S,D_MODEL]
// Block: 256 threads = 8 waves; each wave owns 16 query rows.
// Key dim processed in chunks of KC=64 with online softmax.
// Next chunk's K/V rows are prefetched (global_prefetch_b8).
// =====================================================================
#define KC 64

__global__ __launch_bounds__(256) void attn_wmma_kernel(
    const _Float16* __restrict__ Qp, const _Float16* __restrict__ Kp,
    const _Float16* __restrict__ Vt, const int* __restrict__ mask,
    _Float16* __restrict__ Ows)
{
    __shared__ __align__(16) float    s_sc[8][16][KC + 4];   // 34816 B
    __shared__ __align__(16) _Float16 s_p [8][16][KC + 8];   // 18432 B
    __shared__ float s_alpha[8][16];
    __shared__ float s_linv [8][16];

    const int bh = blockIdx.y;
    const int b  = bh >> 4;
    const int h  = bh & 15;
    const int qbase = blockIdx.x * 128;
    const int tid  = threadIdx.x;
    const int w    = tid >> 5;
    const int lane = tid & 31;
    const int lm   = lane & 15;
    const int sel  = lane >> 4;

    const _Float16* Qb = Qp + (size_t)bh * SEQ * DK;
    const _Float16* Kb = Kp + (size_t)bh * SEQ * DK;
    const _Float16* Vb = Vt + (size_t)bh * DK * SEQ;
    const int* mb_row  = mask + (size_t)b * SEQ;

    // Q fragments for this wave's 16 rows (d split into two 32-chunks)
    F16x16 qf[2];
    {
        const int qrow = qbase + w * 16 + lm;
        const _Float16* p = Qb + (size_t)qrow * DK;
#pragma unroll
        for (int j = 0; j < 2; ++j) {
            qf[j].q[0] = *(const u4*)(p + j * 32 + sel * 8);
            qf[j].q[1] = *(const u4*)(p + j * 32 + 16 + sel * 8);
        }
    }

    VF8 o[4];
#pragma unroll
    for (int fd = 0; fd < 4; ++fd)
#pragma unroll
        for (int r = 0; r < 8; ++r) o[fd].f[r] = 0.0f;

    float m_i = -3.0e38f, l_i = 0.0f;
    const int srow  = lm;                 // softmax row owned by this lane
    const int chalf = sel * (KC / 2);     // this lane's column half
    const float scale = 0.125f;           // 1/sqrt(DK)

    for (int kb = 0; kb < SEQ; kb += KC) {
        // ---- prefetch next chunk's K/V rows into cache (global_prefetch_b8)
        if (kb + KC < SEQ) {
#pragma unroll
            for (int fn = 0; fn < 4; ++fn)
                __builtin_prefetch(Kb + (size_t)(kb + KC + fn * 16 + lm) * DK, 0, 1);
#pragma unroll
            for (int fd = 0; fd < 4; ++fd)
                __builtin_prefetch(Vb + (size_t)(fd * 16 + lm) * SEQ + kb + KC, 0, 1);
        }

        // ---- S = Q K^T * scale + maskbias ----
#pragma unroll
        for (int fn = 0; fn < 4; ++fn) {
            const int n = kb + fn * 16 + lm;          // key position (B-frag col)
            const _Float16* kp = Kb + (size_t)n * DK; // contiguous d-run = B layout
            VF8 s;
#pragma unroll
            for (int r = 0; r < 8; ++r) s.f[r] = 0.0f;
#pragma unroll
            for (int ds = 0; ds < 2; ++ds) {
                F16x16 bfr;
                bfr.q[0] = *(const u4*)(kp + ds * 32 + sel * 8);
                bfr.q[1] = *(const u4*)(kp + ds * 32 + 16 + sel * 8);
                s.v = wmma_f16(qf[ds].v, bfr.v, s.v);
            }
            const float mbias = mb_row[n] ? 0.0f : -1.0e9f;
#pragma unroll
            for (int r = 0; r < 8; ++r)
                s_sc[w][r + 8 * sel][fn * 16 + lm] = s.f[r] * scale + mbias;
        }
        __syncthreads();

        // ---- online softmax: 2 lanes per row, KC/2 cols each ----
        float cmax = -3.0e38f;
#pragma unroll
        for (int c = 0; c < KC / 2; ++c)
            cmax = fmaxf(cmax, s_sc[w][srow][chalf + c]);
        cmax = fmaxf(cmax, __shfl_xor(cmax, 16, 32));
        const float mnew  = fmaxf(m_i, cmax);
        const float alpha = __expf(m_i - mnew);
        float psum = 0.0f;
#pragma unroll
        for (int c = 0; c < KC / 2; ++c) {
            const float p = __expf(s_sc[w][srow][chalf + c] - mnew);
            s_p[w][srow][chalf + c] = (_Float16)p;
            psum += p;
        }
        psum += __shfl_xor(psum, 16, 32);
        l_i = l_i * alpha + psum;
        m_i = mnew;
        if (lane < 16) s_alpha[w][lm] = alpha;
        __syncthreads();

        // ---- rescale O accumulators by per-row alpha ----
        float ar[8];
#pragma unroll
        for (int r = 0; r < 8; ++r) ar[r] = s_alpha[w][r + 8 * sel];
#pragma unroll
        for (int fd = 0; fd < 4; ++fd)
#pragma unroll
            for (int r = 0; r < 8; ++r) o[fd].f[r] *= ar[r];

        // ---- O += P V : A = P from LDS, B = V^T contiguous from global ----
        F16x16 pf[2];
#pragma unroll
        for (int ks = 0; ks < 2; ++ks) {
            const _Float16* pp = &s_p[w][lm][ks * 32];
            pf[ks].q[0] = *(const u4*)(pp + sel * 8);
            pf[ks].q[1] = *(const u4*)(pp + 16 + sel * 8);
        }
#pragma unroll
        for (int fd = 0; fd < 4; ++fd) {
            const int dd = fd * 16 + lm;              // output d col (B-frag col)
            const _Float16* vp = Vb + (size_t)dd * SEQ + kb;
#pragma unroll
            for (int ks = 0; ks < 2; ++ks) {
                F16x16 bfr;
                bfr.q[0] = *(const u4*)(vp + ks * 32 + sel * 8);
                bfr.q[1] = *(const u4*)(vp + ks * 32 + 16 + sel * 8);
                o[fd].v = wmma_f16(pf[ks].v, bfr.v, o[fd].v);
            }
        }
        __syncthreads();   // protect s_p / s_sc for next chunk
    }

    if (lane < 16) s_linv[w][lm] = 1.0f / l_i;
    __syncthreads();
    float lr[8];
#pragma unroll
    for (int r = 0; r < 8; ++r) lr[r] = s_linv[w][r + 8 * sel];

    // ---- store O (f16) to [B,S,D_MODEL]; lanes 0..15 are contiguous cols ----
#pragma unroll
    for (int fd = 0; fd < 4; ++fd) {
        const int dcol = h * DK + fd * 16 + lm;
#pragma unroll
        for (int r = 0; r < 8; ++r) {
            const int qrow = qbase + w * 16 + r + 8 * sel;
            Ows[((size_t)b * SEQ + qrow) * D_MODEL + dcol] =
                (_Float16)(o[fd].f[r] * lr[r]);
        }
    }
}

// =====================================================================
extern "C" void kernel_launch(void* const* d_in, const int* in_sizes, int n_in,
                              void* d_out, int out_size, void* d_ws, size_t ws_size,
                              hipStream_t stream) {
    const float* query = (const float*)d_in[0];
    const float* key_  = (const float*)d_in[1];
    const float* value = (const float*)d_in[2];
    const int*   mask  = (const int*)  d_in[3];
    const float* Wq = (const float*)d_in[4];
    const float* bq = (const float*)d_in[5];
    const float* Wk = (const float*)d_in[6];
    const float* bk = (const float*)d_in[7];
    const float* Wv = (const float*)d_in[8];
    const float* bv = (const float*)d_in[9];
    const float* Wo = (const float*)d_in[10];
    const float* bo = (const float*)d_in[11];

    const size_t TENS = (size_t)BATCH * SEQ * D_MODEL;   // 8192*1024 elements
    _Float16* Qp  = (_Float16*)d_ws;       // [B,H,S,DK]  16 MB
    _Float16* Kp  = Qp  + TENS;            // [B,H,S,DK]  16 MB
    _Float16* Vt  = Kp  + TENS;            // [B,H,DK,S]  16 MB
    _Float16* Ows = Vt  + TENS;            // [B,S,D]     16 MB

    dim3 gg(64, 8, 1), bb(256, 1, 1);
    gemm_wmma_kernel<0, 0><<<gg, bb, 0, stream>>>(query, Wq, bq, (void*)Qp);
    gemm_wmma_kernel<0, 0><<<gg, bb, 0, stream>>>(key_,  Wk, bk, (void*)Kp);
    gemm_wmma_kernel<0, 1><<<gg, bb, 0, stream>>>(value, Wv, bv, (void*)Vt);
    attn_wmma_kernel<<<dim3(16, 64, 1), bb, 0, stream>>>(Qp, Kp, Vt, mask, Ows);
    gemm_wmma_kernel<1, 2><<<gg, bb, 0, stream>>>(Ows, Wo, bo, d_out);
}
